// Qwen3VLVisionAttention_13932873908494
// MI455X (gfx1250) — compile-verified, matrix operand order checked by reading, not run on hardware
//
#include <hip/hip_runtime.h>

#define S_LEN 2048
#define EMBED 1280
#define HEADS 16
#define HDIM  80
#define HDP   96      // head dim padded to multiple of 32 for clean K loops
#define BATCH 2

typedef __attribute__((ext_vector_type(16))) __bf16 v16bf;
typedef __attribute__((ext_vector_type(8)))  float  v8f;
typedef __attribute__((ext_vector_type(4)))  float  vf4;   // native vector for NT stores
typedef __attribute__((ext_vector_type(4)))  unsigned int vu4;
typedef __attribute__((ext_vector_type(8)))  int    vi8;
typedef __attribute__((ext_vector_type(4)))  int    vi4;

union Frag16 { v16bf v; uint4 q[2]; };
union AccF   { v8f  v; float f[8]; };

#if __has_builtin(__builtin_amdgcn_tensor_load_to_lds) && \
    __has_builtin(__builtin_amdgcn_s_wait_tensorcnt)
#define USE_TDM 1
#else
#define USE_TDM 0
#endif

__device__ __forceinline__ unsigned short f2bf(float x) {
  unsigned int u = __float_as_uint(x);
  u += 0x7FFFu + ((u >> 16) & 1u);          // round-to-nearest-even
  return (unsigned short)(u >> 16);
}

#if USE_TDM
// Issue one TDM 2D tile load: rows x 32 bf16 elements from global (row stride
// src_stride elements) into LDS at byte offset lds_byte. LDS pad feature
// reproduces the 40-short row stride: every 16 dwords (pad_interval=3) insert
// 4 dwords (pad_amount=3)  ->  64B row + 16B pad = 80B = 40 shorts.
__device__ __forceinline__ void tdm_load_tile_2d(unsigned lds_byte,
                                                 const unsigned short* gsrc,
                                                 unsigned rows,
                                                 unsigned src_stride) {
  unsigned long long ga = (unsigned long long)(size_t)gsrc;
  vu4 g0;
  g0[0] = 1u;                                        // count=1, user descriptor
  g0[1] = lds_byte;                                  // lds_addr (LDS byte offset)
  g0[2] = (unsigned)ga;                              // global_addr[31:0]
  g0[3] = (unsigned)((ga >> 32) & 0x01FFFFFFu) | (2u << 30);  // addr[56:32], type=2
  const unsigned tdim0 = 1u << 20, tdim1 = 1u << 20; // huge tensor: no OOB clip
  const unsigned tile0 = 32u, tile1 = rows;
  vi8 g1;
  g1[0] = (int)((1u << 16)        // data_size = 1 (2 bytes)
              | (1u << 20)        // pad_enable
              | (3u << 22)        // pad_interval: 16 dwords
              | (3u << 25));      // pad_amount: 4 dwords
  g1[1] = (int)((tdim0 & 0xFFFFu) << 16);                    // tensor_dim0 lo
  g1[2] = (int)((tdim0 >> 16) | ((tdim1 & 0xFFFFu) << 16));  // dim0 hi | dim1 lo
  g1[3] = (int)((tdim1 >> 16) | (tile0 << 16));              // dim1 hi | tile_dim0
  g1[4] = (int)tile1;                                        // tile_dim1, tile_dim2=0
  g1[5] = (int)src_stride;                                   // tensor_dim0_stride lo
  g1[6] = 0;                                                 // stride hi | dim1_stride lo
  g1[7] = 0;
  vi4 z4 = {0, 0, 0, 0};
  vi8 z8 = {0, 0, 0, 0, 0, 0, 0, 0};
  __builtin_amdgcn_tensor_load_to_lds(g0, g1, z4, z4, z8, 0);
}
#endif

// Gather one 16x32 bf16 WMMA operand fragment from an LDS tile row.
// Per ISA 7.12.2: lanes 0-15 hold K=[0..7] (VGPR0-3) and K=[16..23] (VGPR4-7),
// lanes 16-31 hold K=[8..15] and K=[24..31]. Row stride = 40 shorts (80B, 16B aligned).
__device__ __forceinline__ v16bf load_frag(const unsigned short* rowbase, int hi) {
  Frag16 f;
  f.q[0] = *(const uint4*)(rowbase + hi * 8);
  f.q[1] = *(const uint4*)(rowbase + hi * 8 + 16);
  return f.v;
}

__device__ __forceinline__ v8f wmma_bf16(v16bf a, v16bf b, v8f c) {
  return __builtin_amdgcn_wmma_f32_16x16x32_bf16(false, a, false, b, (short)0, c,
                                                 false, false);
}

// ---------------------------------------------------------------- zero fill
__global__ void zero_ws_kernel(uint4* __restrict__ p, int n4) {
  uint4 z = {0u, 0u, 0u, 0u};
  for (int i = blockIdx.x * blockDim.x + threadIdx.x; i < n4;
       i += gridDim.x * blockDim.x)
    p[i] = z;
}

// --------------------------------------------------- QKV GEMM (fp32 inputs)
// C[4096,3840] = H[4096,1280] x Wqkv[3840,1280]^T + b ; scatter to Q/K/Vt bf16
__global__ __launch_bounds__(256) void qkv_gemm_kernel(
    const float* __restrict__ H, const float* __restrict__ W,
    const float* __restrict__ bias,
    unsigned short* __restrict__ Qbf, unsigned short* __restrict__ Kbf,
    unsigned short* __restrict__ Vt) {
  __shared__ __align__(16) unsigned short As[64 * 40];
  __shared__ __align__(16) unsigned short Bs[128 * 40];
  const int m0 = blockIdx.x * 64, n0 = blockIdx.y * 128;
  const int lane = threadIdx.x & 31, lr = lane & 15, hi = lane >> 4;
  const int w = threadIdx.x >> 5, wm = w >> 2, wn = w & 3;
  const v8f vzero = {0.f, 0.f, 0.f, 0.f, 0.f, 0.f, 0.f, 0.f};
  v8f acc[2][2];
  acc[0][0] = vzero; acc[0][1] = vzero; acc[1][0] = vzero; acc[1][1] = vzero;

  for (int kt = 0; kt < EMBED; kt += 32) {
    for (int g = threadIdx.x; g < 512; g += 256) {     // A: 64x32 fp32 -> bf16
      int row = g >> 3, col = (g & 7) << 2;
      float4 f = *(const float4*)(H + (size_t)(m0 + row) * EMBED + kt + col);
      unsigned short* d = &As[row * 40 + col];
      d[0] = f2bf(f.x); d[1] = f2bf(f.y); d[2] = f2bf(f.z); d[3] = f2bf(f.w);
    }
    for (int g = threadIdx.x; g < 1024; g += 256) {    // B: 128x32 fp32 -> bf16
      int row = g >> 3, col = (g & 7) << 2;
      float4 f = *(const float4*)(W + (size_t)(n0 + row) * EMBED + kt + col);
      unsigned short* d = &Bs[row * 40 + col];
      d[0] = f2bf(f.x); d[1] = f2bf(f.y); d[2] = f2bf(f.z); d[3] = f2bf(f.w);
    }
    __syncthreads();
    v16bf a0 = load_frag(&As[(wm * 32 + lr) * 40], hi);
    v16bf a1 = load_frag(&As[(wm * 32 + 16 + lr) * 40], hi);
    v16bf b0 = load_frag(&Bs[(wn * 32 + lr) * 40], hi);
    v16bf b1 = load_frag(&Bs[(wn * 32 + 16 + lr) * 40], hi);
    acc[0][0] = wmma_bf16(a0, b0, acc[0][0]);
    acc[0][1] = wmma_bf16(a0, b1, acc[0][1]);
    acc[1][0] = wmma_bf16(a1, b0, acc[1][0]);
    acc[1][1] = wmma_bf16(a1, b1, acc[1][1]);
    __syncthreads();
  }
#pragma unroll
  for (int i = 0; i < 2; ++i)
#pragma unroll
    for (int j = 0; j < 2; ++j) {
      AccF r; r.v = acc[i][j];
#pragma unroll
      for (int e = 0; e < 8; ++e) {
        int gm = m0 + wm * 32 + i * 16 + hi * 8 + e;   // token row
        int gn = n0 + wn * 32 + j * 16 + lr;           // qkv column
        float val = r.f[e] + bias[gn];
        int part = gn / EMBED, c = gn - part * EMBED;
        int h = c / HDIM, d = c - h * HDIM;
        int b = gm >> 11, s = gm & (S_LEN - 1);
        int bh = b * HEADS + h;
        unsigned short bv = f2bf(val);
        if (part == 0)      Qbf[((size_t)bh * S_LEN + s) * HDP + d] = bv;
        else if (part == 1) Kbf[((size_t)bh * S_LEN + s) * HDP + d] = bv;
        else                Vt [((size_t)bh * HDP + d) * S_LEN + s] = bv;
      }
    }
}

// -------------------------------------------- logits: scale * Q K^T (bf16 in)
__global__ __launch_bounds__(256) void logits_gemm_kernel(
    const unsigned short* __restrict__ Qbf, const unsigned short* __restrict__ Kbf,
    float* __restrict__ attn) {
  __shared__ __align__(16) unsigned short As[64 * 40];
  __shared__ __align__(16) unsigned short Bs[128 * 40];
  const int m0 = blockIdx.x * 64, n0 = blockIdx.y * 128, bh = blockIdx.z;
  const unsigned short* Q  = Qbf + (size_t)bh * S_LEN * HDP;
  const unsigned short* Kp = Kbf + (size_t)bh * S_LEN * HDP;
  float* Ab = attn + (size_t)bh * S_LEN * S_LEN;
  const int lane = threadIdx.x & 31, lr = lane & 15, hi = lane >> 4;
  const int w = threadIdx.x >> 5, wm = w >> 2, wn = w & 3;
  const v8f vzero = {0.f, 0.f, 0.f, 0.f, 0.f, 0.f, 0.f, 0.f};
  v8f acc[2][2];
  acc[0][0] = vzero; acc[0][1] = vzero; acc[1][0] = vzero; acc[1][1] = vzero;
#if USE_TDM
  const unsigned asAddr = (unsigned)(size_t)(void*)As;
  const unsigned bsAddr = (unsigned)(size_t)(void*)Bs;
#endif

  for (int kt = 0; kt < HDP; kt += 32) {
#if USE_TDM
    if (w == 0) {                                      // TDM DMA both tiles
      tdm_load_tile_2d(asAddr, Q + (size_t)m0 * HDP + kt, 64, HDP);
      tdm_load_tile_2d(bsAddr, Kp + (size_t)n0 * HDP + kt, 128, HDP);
      __builtin_amdgcn_s_wait_tensorcnt(0);
    }
#else
    for (int g = threadIdx.x; g < 256; g += 256) {     // A: 64x32 bf16 copy
      int row = g >> 2, col = (g & 3) << 3;
      *(uint4*)&As[row * 40 + col] =
          *(const uint4*)(Q + (size_t)(m0 + row) * HDP + kt + col);
    }
    for (int g = threadIdx.x; g < 512; g += 256) {     // B: 128x32 bf16 copy
      int row = g >> 2, col = (g & 3) << 3;
      *(uint4*)&Bs[row * 40 + col] =
          *(const uint4*)(Kp + (size_t)(n0 + row) * HDP + kt + col);
    }
#endif
    __syncthreads();
    v16bf a0 = load_frag(&As[(wm * 32 + lr) * 40], hi);
    v16bf a1 = load_frag(&As[(wm * 32 + 16 + lr) * 40], hi);
    v16bf b0 = load_frag(&Bs[(wn * 32 + lr) * 40], hi);
    v16bf b1 = load_frag(&Bs[(wn * 32 + 16 + lr) * 40], hi);
    acc[0][0] = wmma_bf16(a0, b0, acc[0][0]);
    acc[0][1] = wmma_bf16(a0, b1, acc[0][1]);
    acc[1][0] = wmma_bf16(a1, b0, acc[1][0]);
    acc[1][1] = wmma_bf16(a1, b1, acc[1][1]);
    __syncthreads();
  }
  const float scale = 0.11180339887498948f;            // 1/sqrt(80)
#pragma unroll
  for (int i = 0; i < 2; ++i)
#pragma unroll
    for (int j = 0; j < 2; ++j) {
      AccF r; r.v = acc[i][j];
#pragma unroll
      for (int e = 0; e < 8; ++e) {
        int gm = m0 + wm * 32 + i * 16 + hi * 8 + e;
        int gn = n0 + wn * 32 + j * 16 + lr;
        __builtin_nontemporal_store(r.f[e] * scale, &Ab[(size_t)gm * S_LEN + gn]);
      }
    }
}

// ------------------------------------------------------- row softmax in-place
__global__ __launch_bounds__(256) void softmax_kernel(float* __restrict__ attn) {
  float* p = attn + (size_t)blockIdx.x * S_LEN;
  const int t = threadIdx.x;
  vf4 x0 = ((const vf4*)p)[t];
  vf4 x1 = ((const vf4*)p)[t + 256];
  float m = fmaxf(fmaxf(fmaxf(x0.x, x0.y), fmaxf(x0.z, x0.w)),
                  fmaxf(fmaxf(x1.x, x1.y), fmaxf(x1.z, x1.w)));
#pragma unroll
  for (int d = 16; d; d >>= 1) m = fmaxf(m, __shfl_xor(m, d, 32));
  __shared__ float redm[8];
  __shared__ float reds[8];
  if ((t & 31) == 0) redm[t >> 5] = m;
  __syncthreads();
  m = redm[0];
#pragma unroll
  for (int i = 1; i < 8; ++i) m = fmaxf(m, redm[i]);
  x0.x = __expf(x0.x - m); x0.y = __expf(x0.y - m);
  x0.z = __expf(x0.z - m); x0.w = __expf(x0.w - m);
  x1.x = __expf(x1.x - m); x1.y = __expf(x1.y - m);
  x1.z = __expf(x1.z - m); x1.w = __expf(x1.w - m);
  float s = x0.x + x0.y + x0.z + x0.w + x1.x + x1.y + x1.z + x1.w;
#pragma unroll
  for (int d = 16; d; d >>= 1) s += __shfl_xor(s, d, 32);
  if ((t & 31) == 0) reds[t >> 5] = s;
  __syncthreads();
  s = reds[0] + reds[1] + reds[2] + reds[3] + reds[4] + reds[5] + reds[6] + reds[7];
  float inv = 1.0f / s;
  x0 *= inv;
  x1 *= inv;
  __builtin_nontemporal_store(x0, (vf4*)p + t);
  __builtin_nontemporal_store(x1, (vf4*)p + t + 256);
}

// ---------------------------------------------- ctx = attn x V (N = 96 padded)
__global__ __launch_bounds__(192) void attnv_gemm_kernel(
    const float* __restrict__ attn, const unsigned short* __restrict__ Vt,
    unsigned short* __restrict__ ctx) {
  __shared__ __align__(16) unsigned short As[64 * 40];
  __shared__ __align__(16) unsigned short Bs[96 * 40];
  const int m0 = blockIdx.x * 64, bh = blockIdx.z;
  const int b = bh >> 4, h = bh & 15;
  const float* Ap = attn + (size_t)bh * S_LEN * S_LEN;
  const unsigned short* Bp = Vt + (size_t)bh * HDP * S_LEN;
  const int lane = threadIdx.x & 31, lr = lane & 15, hi = lane >> 4;
  const int w = threadIdx.x >> 5, wm = w / 3, wn = w - wm * 3;   // 2x3 waves
  const v8f vzero = {0.f, 0.f, 0.f, 0.f, 0.f, 0.f, 0.f, 0.f};
  v8f acc[2][2];
  acc[0][0] = vzero; acc[0][1] = vzero; acc[1][0] = vzero; acc[1][1] = vzero;
#if USE_TDM
  const unsigned bsAddr = (unsigned)(size_t)(void*)Bs;
#endif

  for (int kt = 0; kt < S_LEN; kt += 32) {
#if USE_TDM
    if (w == 0)                                        // B tile via TDM (async)
      tdm_load_tile_2d(bsAddr, Bp + kt, 96, S_LEN);
#else
    for (int g = threadIdx.x; g < 384; g += 192) {     // B: 96x32 bf16 copy
      int row = g >> 2, col = (g & 3) << 3;
      *(uint4*)&Bs[row * 40 + col] =
          *(const uint4*)(Bp + (size_t)row * S_LEN + kt + col);
    }
#endif
    for (int g = threadIdx.x; g < 512; g += 192) {     // A: 64x32 fp32 -> bf16
      int row = g >> 3, col = (g & 7) << 2;
      float4 f = *(const float4*)(Ap + (size_t)(m0 + row) * S_LEN + kt + col);
      unsigned short* d = &As[row * 40 + col];
      d[0] = f2bf(f.x); d[1] = f2bf(f.y); d[2] = f2bf(f.z); d[3] = f2bf(f.w);
    }
#if USE_TDM
    if (w == 0) __builtin_amdgcn_s_wait_tensorcnt(0);
#endif
    __syncthreads();
    v16bf a0 = load_frag(&As[(wm * 32 + lr) * 40], hi);
    v16bf a1 = load_frag(&As[(wm * 32 + 16 + lr) * 40], hi);
    v16bf b0 = load_frag(&Bs[(wn * 32 + lr) * 40], hi);
    v16bf b1 = load_frag(&Bs[(wn * 32 + 16 + lr) * 40], hi);
    acc[0][0] = wmma_bf16(a0, b0, acc[0][0]);
    acc[0][1] = wmma_bf16(a0, b1, acc[0][1]);
    acc[1][0] = wmma_bf16(a1, b0, acc[1][0]);
    acc[1][1] = wmma_bf16(a1, b1, acc[1][1]);
    __syncthreads();
  }
#pragma unroll
  for (int i = 0; i < 2; ++i)
#pragma unroll
    for (int j = 0; j < 2; ++j) {
      AccF r; r.v = acc[i][j];
#pragma unroll
      for (int e = 0; e < 8; ++e) {
        int gm = m0 + wm * 32 + i * 16 + hi * 8 + e;   // q index
        int gn = wn * 32 + j * 16 + lr;                // d index (0..95)
        if (gn < HDIM)
          ctx[((size_t)b * S_LEN + gm) * EMBED + h * HDIM + gn] = f2bf(r.f[e]);
      }
    }
}

// -------------------------------------------------- output projection + bias
__global__ __launch_bounds__(256) void proj_gemm_kernel(
    const unsigned short* __restrict__ ctx, const float* __restrict__ W,
    const float* __restrict__ bias, float* __restrict__ out) {
  __shared__ __align__(16) unsigned short As[64 * 40];
  __shared__ __align__(16) unsigned short Bs[128 * 40];
  const int m0 = blockIdx.x * 64, n0 = blockIdx.y * 128;
  const int lane = threadIdx.x & 31, lr = lane & 15, hi = lane >> 4;
  const int w = threadIdx.x >> 5, wm = w >> 2, wn = w & 3;
  const v8f vzero = {0.f, 0.f, 0.f, 0.f, 0.f, 0.f, 0.f, 0.f};
  v8f acc[2][2];
  acc[0][0] = vzero; acc[0][1] = vzero; acc[1][0] = vzero; acc[1][1] = vzero;
#if USE_TDM
  const unsigned asAddr = (unsigned)(size_t)(void*)As;
#endif

  for (int kt = 0; kt < EMBED; kt += 32) {
#if USE_TDM
    if (w == 0)                                        // A tile via TDM (async)
      tdm_load_tile_2d(asAddr, ctx + (size_t)m0 * EMBED + kt, 64, EMBED);
#else
    for (int g = threadIdx.x; g < 256; g += 256) {     // A: 64x32 bf16 copy
      int row = g >> 2, col = (g & 3) << 3;
      *(uint4*)&As[row * 40 + col] =
          *(const uint4*)(ctx + (size_t)(m0 + row) * EMBED + kt + col);
    }
#endif
    for (int g = threadIdx.x; g < 1024; g += 256) {    // B: 128x32 fp32 -> bf16
      int row = g >> 3, col = (g & 7) << 2;
      float4 f = *(const float4*)(W + (size_t)(n0 + row) * EMBED + kt + col);
      unsigned short* d = &Bs[row * 40 + col];
      d[0] = f2bf(f.x); d[1] = f2bf(f.y); d[2] = f2bf(f.z); d[3] = f2bf(f.w);
    }
#if USE_TDM
    if (w == 0) __builtin_amdgcn_s_wait_tensorcnt(0);
#endif
    __syncthreads();
    v16bf a0 = load_frag(&As[(wm * 32 + lr) * 40], hi);
    v16bf a1 = load_frag(&As[(wm * 32 + 16 + lr) * 40], hi);
    v16bf b0 = load_frag(&Bs[(wn * 32 + lr) * 40], hi);
    v16bf b1 = load_frag(&Bs[(wn * 32 + 16 + lr) * 40], hi);
    acc[0][0] = wmma_bf16(a0, b0, acc[0][0]);
    acc[0][1] = wmma_bf16(a0, b1, acc[0][1]);
    acc[1][0] = wmma_bf16(a1, b0, acc[1][0]);
    acc[1][1] = wmma_bf16(a1, b1, acc[1][1]);
    __syncthreads();
  }
#pragma unroll
  for (int i = 0; i < 2; ++i)
#pragma unroll
    for (int j = 0; j < 2; ++j) {
      AccF r; r.v = acc[i][j];
#pragma unroll
      for (int e = 0; e < 8; ++e) {
        int gm = m0 + wm * 32 + i * 16 + hi * 8 + e;
        int gn = n0 + wn * 32 + j * 16 + lr;
        out[(size_t)gm * EMBED + gn] = r.f[e] + bias[gn];
      }
    }
}

extern "C" void kernel_launch(void* const* d_in, const int* in_sizes, int n_in,
                              void* d_out, int out_size, void* d_ws, size_t ws_size,
                              hipStream_t stream) {
  const float* hidden = (const float*)d_in[0];   // [2,2048,1280]
  const float* w_qkv  = (const float*)d_in[1];   // [3840,1280]
  const float* b_qkv  = (const float*)d_in[2];   // [3840]
  const float* w_proj = (const float*)d_in[3];   // [1280,1280]
  const float* b_proj = (const float*)d_in[4];   // [1280]

  float* out  = (float*)d_out;                               // [2,2048,1280]
  float* attn = out + (size_t)BATCH * S_LEN * EMBED;         // [2,16,2048,2048]

  const size_t qkv_elems = (size_t)BATCH * HEADS * S_LEN * HDP; // 6,291,456
  unsigned short* Qbf = (unsigned short*)d_ws;
  unsigned short* Kbf = Qbf + qkv_elems;
  unsigned short* Vt  = Kbf + qkv_elems;                     // [bh, HDP, S]
  unsigned short* ctx = Vt  + qkv_elems;                     // [4096, 1280] bf16

  // 1) zero Q/K/Vt so the d=80..95 pad contributes nothing
  zero_ws_kernel<<<2048, 256, 0, stream>>>((uint4*)d_ws,
                                           (int)(3 * qkv_elems * 2 / 16));
  // 2) QKV projection
  qkv_gemm_kernel<<<dim3(4096 / 64, 3840 / 128), 256, 0, stream>>>(
      hidden, w_qkv, b_qkv, Qbf, Kbf, Vt);
  // 3) logits = scale * Q K^T
  logits_gemm_kernel<<<dim3(S_LEN / 64, S_LEN / 128, BATCH * HEADS), 256, 0,
                       stream>>>(Qbf, Kbf, attn);
  // 4) softmax over each of the 65536 rows
  softmax_kernel<<<dim3(BATCH * HEADS * S_LEN), 256, 0, stream>>>(attn);
  // 5) ctx = attn x V
  attnv_gemm_kernel<<<dim3(S_LEN / 64, 1, BATCH * HEADS), 192, 0, stream>>>(
      attn, Vt, ctx);
  // 6) out = ctx x Wproj^T + b
  proj_gemm_kernel<<<dim3(4096 / 64, EMBED / 128), 256, 0, stream>>>(
      ctx, w_proj, b_proj, out);
}
